// Recurrence_20993800142907
// MI455X (gfx1250) — compile-verified, matrix operand order checked by reading, not run on hardware
//
#include <hip/hip_runtime.h>
#include <hip/hip_bf16.h>

// ---------------------------------------------------------------------------
// Types for CDNA5 WMMA
// ---------------------------------------------------------------------------
typedef __bf16 bf16;
typedef __attribute__((ext_vector_type(16))) __bf16 v16bf;
typedef __attribute__((ext_vector_type(8)))  float  v8f;

#define EPS 1e-8f

// Problem dims (fixed by reference)
#define TT 128
#define NN 512
#define DOBS 128
#define SS 32
#define DE 256
#define DD 256      /* GRU hidden = 4*DD = 1024 */
#define HH 256
#define AA 64
#define G3 3072     /* 3 * 4D */
#define HID4 1024   /* 4D */
#define KF 384      /* DD + DOBS */

// ---------------------------------------------------------------------------
// CDNA5 async global -> LDS copy (ASYNCcnt path). Inline asm: portable across
// ROCm 7.2 / amdgpu-toolchain (bypasses churn-prone clang builtins).
//   global_load_async_to_lds_b128 vdst(lds byte addr), vaddr(64b), off
// ---------------------------------------------------------------------------
__device__ __forceinline__ void async_cp_b128(const bf16* g, bf16* l) {
    unsigned lds_off = (unsigned)(unsigned long long)(l);
    unsigned long long ga = (unsigned long long)(g);
    asm volatile("global_load_async_to_lds_b128 %0, %1, off"
                 :: "v"(lds_off), "v"(ga) : "memory");
}
__device__ __forceinline__ void async_wait0() {
    asm volatile("s_wait_asynccnt 0x0" ::: "memory");
}

// ---------------------------------------------------------------------------
// WMMA helpers: 16x16x32 bf16, fp32 accumulate
//   A 16x32 (MxK): lane m=lane&15; half = (lane>>4)*8; VGPR i<4 -> K=2i+half,
//   VGPR i>=4 -> K=16+2(i-4)+half; 2 bf16 per VGPR.
//   B 32x16 (KxN): lane owns column n=lane&15, same K packing.
//   C/D: VGPR j -> row j + 8*(lane>>4), col lane&15.
// ---------------------------------------------------------------------------
__device__ __forceinline__ v16bf load_tile_A(const bf16* base, int ld, int lane) {
    // element (m,k) at base[m*ld + k]
    int m  = lane & 15;
    int kh = (lane >> 4) << 3;
    const bf16* row = base + (long)m * ld;
    v16bf r;
#pragma unroll
    for (int i = 0; i < 4; ++i) {
        int k0 = 2 * i + kh;
        r[2 * i]     = row[k0];
        r[2 * i + 1] = row[k0 + 1];
        int k1 = 16 + 2 * i + kh;
        r[8 + 2 * i]     = row[k1];
        r[8 + 2 * i + 1] = row[k1 + 1];
    }
    return r;
}

__device__ __forceinline__ v16bf load_tile_B_kcontig(const bf16* base, int ld, int lane) {
    // element (k,n) at base[n*ld + k]  (k contiguous within a column)
    int n  = lane & 15;
    int kh = (lane >> 4) << 3;
    const bf16* col = base + (long)n * ld;
    v16bf r;
#pragma unroll
    for (int i = 0; i < 4; ++i) {
        int k0 = 2 * i + kh;
        r[2 * i]     = col[k0];
        r[2 * i + 1] = col[k0 + 1];
        int k1 = 16 + 2 * i + kh;
        r[8 + 2 * i]     = col[k1];
        r[8 + 2 * i + 1] = col[k1 + 1];
    }
    return r;
}

__device__ __forceinline__ v16bf load_tile_B_krow(const bf16* base, int ld, int lane) {
    // element (k,n) at base[k*ld + n]  (row-major over k)
    int n  = lane & 15;
    int kh = (lane >> 4) << 3;
    v16bf r;
#pragma unroll
    for (int i = 0; i < 4; ++i) {
        int k0 = 2 * i + kh;
        r[2 * i]     = base[(long)k0 * ld + n];
        r[2 * i + 1] = base[(long)(k0 + 1) * ld + n];
        int k1 = 16 + 2 * i + kh;
        r[8 + 2 * i]     = base[(long)k1 * ld + n];
        r[8 + 2 * i + 1] = base[(long)(k1 + 1) * ld + n];
    }
    return r;
}

__device__ __forceinline__ v8f wmma_bf16(v16bf a, v16bf b, v8f c) {
    return __builtin_amdgcn_wmma_f32_16x16x32_bf16(false, a, false, b, (short)0, c,
                                                   false, false);
}

// ---------------------------------------------------------------------------
// Utility kernels
// ---------------------------------------------------------------------------
__global__ __launch_bounds__(256) void cast_f32_bf16_kernel(const float* __restrict__ src,
                                                            bf16* __restrict__ dst,
                                                            long n) {
    long i = (long)blockIdx.x * 256 + threadIdx.x;
    if (i < n) dst[i] = (bf16)src[i];
}

__global__ __launch_bounds__(256) void gather_embed_kernel(const float* __restrict__ embed,
                                                           const int* __restrict__ subt,
                                                           bf16* __restrict__ out) {
    long i = (long)blockIdx.x * 256 + threadIdx.x;   // over N*S*De
    if (i >= (long)NN * SS * DE) return;
    int d  = (int)(i & (DE - 1));
    long ns = i >> 8;                                // n*S + s
    int row = subt[ns];
    out[i] = (bf16)embed[(long)row * DE + d];
}

// ---------------------------------------------------------------------------
// GRU step, pass 1: fused GEMM with async-LDS staging + 2D block tiling.
//   Block tile: 128 rows (batch) x 64 cols (gate units), 8 waves.
//   Wave sub-tile: 32x32 = 2x2 WMMA fragments.
//   Per 32-wide K chunk: stage A[128x32], B[64x32] into LDS via
//   global_load_async_to_lds_b128, then 4 v_wmma per wave from LDS.
//   ax = x @ W_ih.T (K=256), ah = h @ W_hh.T (K=1024, skipped when s==0).
//   j <  2048 : prz = ax+ah ; j >= 2048 : gin = ax, ghn = ah (n-gate split).
// Grid: (512/128) * (3072/64) = 192 blocks x 256 threads.
// ---------------------------------------------------------------------------
#define BM 128
#define BN 64
#define LDPAD 8
__global__ __launch_bounds__(256) void gru_gemm_kernel(
    const bf16* __restrict__ emb, const bf16* __restrict__ hbf,
    const bf16* __restrict__ wih, const bf16* __restrict__ whh,
    float* __restrict__ prz, float* __restrict__ gin, float* __restrict__ ghn,
    int s) {
    __shared__ __align__(16) bf16 sA[BM][32 + LDPAD];   // 10 KB
    __shared__ __align__(16) bf16 sB[BN][32 + LDPAD];   //  5 KB
    const int tid  = threadIdx.x;
    const int wave = tid >> 5, lane = tid & 31;
    const int bm = blockIdx.x & 3;          // 4 row blocks
    const int bj = blockIdx.x >> 2;         // 48 col blocks
    const int m0 = bm * BM;
    const int j0 = bj * BN;
    const int wr = wave & 3;                // wave rows:  wr*32
    const int wc = wave >> 2;               // wave cols:  wc*32

    v8f accx[2][2], acch[2][2];
#pragma unroll
    for (int i = 0; i < 2; ++i)
#pragma unroll
        for (int j = 0; j < 2; ++j) {
            accx[i][j] = (v8f){0.f, 0.f, 0.f, 0.f, 0.f, 0.f, 0.f, 0.f};
            acch[i][j] = (v8f){0.f, 0.f, 0.f, 0.f, 0.f, 0.f, 0.f, 0.f};
        }

    // Stage one K-chunk (32 wide) into LDS asynchronously and issue 4 WMMAs.
    auto chunk = [&](const bf16* arow0, long lda, const bf16* bcol0, long ldb,
                     int k0, v8f (&acc)[2][2]) {
        // A: 128 rows x 64B -> 512 x b128 segments, 2 per thread
#pragma unroll
        for (int i = 0; i < 2; ++i) {
            int idx = tid + i * 256;
            int r = idx >> 2, seg = idx & 3;
            async_cp_b128(arow0 + (long)r * lda + k0 + seg * 8, &sA[r][seg * 8]);
        }
        // B: 64 cols x 64B -> 256 x b128 segments, 1 per thread
        {
            int c = tid >> 2, seg = tid & 3;
            async_cp_b128(bcol0 + (long)c * ldb + k0 + seg * 8, &sB[c][seg * 8]);
        }
        async_wait0();
        __syncthreads();
        v16bf a0 = load_tile_A(&sA[wr * 32][0],      32 + LDPAD, lane);
        v16bf a1 = load_tile_A(&sA[wr * 32 + 16][0], 32 + LDPAD, lane);
        v16bf b0 = load_tile_B_kcontig(&sB[wc * 32][0],      32 + LDPAD, lane);
        v16bf b1 = load_tile_B_kcontig(&sB[wc * 32 + 16][0], 32 + LDPAD, lane);
        acc[0][0] = wmma_bf16(a0, b0, acc[0][0]);
        acc[0][1] = wmma_bf16(a0, b1, acc[0][1]);
        acc[1][0] = wmma_bf16(a1, b0, acc[1][0]);
        acc[1][1] = wmma_bf16(a1, b1, acc[1][1]);
        __syncthreads();
    };

    // x-contribution: rows of emb at step s (row n -> emb[(n*S+s)*DE])
    {
        const bf16* arow0 = emb + ((long)m0 * SS + s) * DE;
        const bf16* bcol0 = wih + (long)j0 * DE;
        for (int kk = 0; kk < DE / 32; ++kk)
            chunk(arow0, (long)SS * DE, bcol0, DE, kk * 32, accx);
    }
    // h-contribution (h0 == 0 -> skip at s==0)
    if (s > 0) {
        const bf16* arow0 = hbf + (long)m0 * HID4;
        const bf16* bcol0 = whh + (long)j0 * HID4;
        for (int kk = 0; kk < HID4 / 32; ++kk)
            chunk(arow0, HID4, bcol0, HID4, kk * 32, acch);
    }

    const int col = lane & 15, rb = (lane >> 4) << 3;
#pragma unroll
    for (int ti = 0; ti < 2; ++ti) {
#pragma unroll
        for (int tj = 0; tj < 2; ++tj) {
            int jg = j0 + wc * 32 + tj * 16 + col;
#pragma unroll
            for (int j = 0; j < 8; ++j) {
                int n = m0 + wr * 32 + ti * 16 + rb + j;
                float x = accx[ti][tj][j];
                float h = acch[ti][tj][j];
                if (jg < 2048) {
                    prz[(long)n * 2048 + jg] = x + h;
                } else {
                    gin[(long)n * HID4 + jg - 2048] = x;
                    ghn[(long)n * HID4 + jg - 2048] = h;
                }
            }
        }
    }
}

// ---------------------------------------------------------------------------
// GRU step, pass 2: gates + hidden update + scatter into memory banks
//   r = sig(ar), z = sig(az), nn = tanh(i_n + r*h_n), h = (1-z)*nn + z*h
//   Mall[:,s,:] split: M | M_minus | M_plus | p0f (mean -> p0)
// ---------------------------------------------------------------------------
__global__ __launch_bounds__(256) void gru_gate_kernel(
    const float* __restrict__ prz, const float* __restrict__ gin,
    const float* __restrict__ ghn, const float* __restrict__ bih,
    const float* __restrict__ bhh, float* __restrict__ hf, bf16* __restrict__ hb,
    bf16* __restrict__ Mb, bf16* __restrict__ Mmb, bf16* __restrict__ Mpb,
    float* __restrict__ p0, int s) {
    int idx = blockIdx.x * 256 + threadIdx.x;   // N*1024 elems
    int n = idx >> 10, d = idx & 1023;

    float ar  = prz[(long)n * 2048 + d] + bih[d] + bhh[d];
    float az  = prz[(long)n * 2048 + 1024 + d] + bih[1024 + d] + bhh[1024 + d];
    float gi_n = gin[(long)n * HID4 + d] + bih[2048 + d];
    float gh_n = ghn[(long)n * HID4 + d] + bhh[2048 + d];

    float r  = 1.f / (1.f + expf(-ar));
    float z  = 1.f / (1.f + expf(-az));
    float nn = tanhf(gi_n + r * gh_n);
    float hp = (s == 0) ? 0.f : hf[(long)n * HID4 + d];
    float hnew = (1.f - z) * nn + z * hp;

    hf[(long)n * HID4 + d] = hnew;
    hb[(long)n * HID4 + d] = (bf16)hnew;

    long mi = ((long)n * SS + s) * DD;
    if (d < 256)       Mb [mi + d]       = (bf16)hnew;
    else if (d < 512)  Mmb[mi + d - 256] = (bf16)hnew;
    else if (d < 768)  Mpb[mi + d - 512] = (bf16)hnew;

    // p0 = mean of last quarter; each block lies entirely within one quarter
    __shared__ float red[256];
    bool in4 = (d >= 768);
    red[threadIdx.x] = in4 ? hnew : 0.f;
    __syncthreads();
#pragma unroll
    for (int off = 128; off > 0; off >>= 1) {
        if (threadIdx.x < off) red[threadIdx.x] += red[threadIdx.x + off];
        __syncthreads();
    }
    if (threadIdx.x == 0 && in4) p0[n * SS + s] = red[0] * (1.f / 256.f);
}

// ---------------------------------------------------------------------------
// psi: E = tanh(obs @ psi_w + psi_b), [65536,128] x [128,256] -> bf16
// One wave per 16x16 tile: 4096 x 16 tiles, 8 waves/block.
// ---------------------------------------------------------------------------
__global__ __launch_bounds__(256) void psi_kernel(const bf16* __restrict__ obsb,
                                                  const bf16* __restrict__ psiw,
                                                  const float* __restrict__ psib,
                                                  bf16* __restrict__ Eb) {
    int wid  = (blockIdx.x << 3) + (threadIdx.x >> 5);
    int lane = threadIdx.x & 31;
    long tm = wid >> 4;     // 4096 row tiles over T*N
    int  tj = wid & 15;     // 16 col tiles over 256
    long m0 = tm << 4;
    int  j0 = tj << 4;

    v8f acc = {0.f, 0.f, 0.f, 0.f, 0.f, 0.f, 0.f, 0.f};
    const bf16* abase = obsb + m0 * DOBS;
#pragma unroll
    for (int kk = 0; kk < DOBS / 32; ++kk) {
        v16bf a = load_tile_A(abase + kk * 32, DOBS, lane);
        v16bf b = load_tile_B_krow(psiw + (long)(kk * 32) * DD + j0, DD, lane);
        acc = wmma_bf16(a, b, acc);
    }
    int col = lane & 15, rb = (lane >> 4) << 3;
#pragma unroll
    for (int j = 0; j < 8; ++j) {
        long row = m0 + rb + j;
        Eb[row * DD + j0 + col] = (bf16)tanhf(acc[j] + psib[j0 + col]);
    }
}

// Row L2 norms of a bf16 matrix
__global__ __launch_bounds__(256) void rownorm_kernel(const bf16* __restrict__ X,
                                                      float* __restrict__ out,
                                                      int rows, int cols) {
    int r = blockIdx.x * 256 + threadIdx.x;
    if (r >= rows) return;
    const bf16* p = X + (long)r * cols;
    float s = 0.f;
    for (int c = 0; c < cols; ++c) { float v = (float)p[c]; s += v * v; }
    out[r] = sqrtf(s);
}

// p = new_episode ? p0 : p_init (per n)
__global__ __launch_bounds__(256) void init_p_kernel(const float* __restrict__ p_init,
                                                     const float* __restrict__ p0,
                                                     float* __restrict__ p) {
    int n = blockIdx.x * 256 + threadIdx.x;
    if (n >= NN) return;
    bool newep = true;
    for (int s = 0; s < SS; ++s)
        if (p_init[n * SS + s] != 0.f) newep = false;
    for (int s = 0; s < SS; ++s)
        p[n * SS + s] = newep ? p0[n * SS + s] : p_init[n * SS + s];
}

// ---------------------------------------------------------------------------
// The T-scan. Recurrence on p is independent per n -> persistent blocks.
// 32 blocks, each owns 16 consecutive n and loops t = 0..127.
// hid GEMM [16,384]x[384,256] : 16 waves x (16x16 tile, 12 K-steps) of WMMA.
// actor GEMM [16,256]x[256,64]: 4 waves of WMMA.
// ---------------------------------------------------------------------------
#define NL 16
__global__ __launch_bounds__(512) void scan_kernel(
    const bf16* __restrict__ Eb, const float* __restrict__ enorm,
    const bf16* __restrict__ Mb, const bf16* __restrict__ Mpb,
    const bf16* __restrict__ Mmb, const float* __restrict__ mpn,
    const float* __restrict__ mmn, const float* __restrict__ pinit,
    const float* __restrict__ obs, const float* __restrict__ u,
    const bf16* __restrict__ fwb, const float* __restrict__ fb,
    const bf16* __restrict__ awb, const float* __restrict__ ab,
    const float* __restrict__ cw, const float* __restrict__ cb,
    int* __restrict__ a_out, float* __restrict__ probs_out,
    float* __restrict__ v_out) {
    __shared__ float sE[NL][DD];
    __shared__ bf16  sA[NL][KF];
    __shared__ float sP[NL][SS];
    __shared__ bf16  sHb[NL][HH];
    __shared__ float sHid[NL][HH];
    __shared__ float sLog[NL][AA];

    int tid = threadIdx.x;
    int n0  = blockIdx.x * NL;
    int wave = tid >> 5, lane = tid & 31;

    // load p state (NL*SS == 512 == blockDim)
    sP[tid >> 5][tid & 31] = pinit[(n0 + (tid >> 5)) * SS + (tid & 31)];
    __syncthreads();

    for (int t = 0; t < TT; ++t) {
        // stage e (fp32 in LDS)
        for (int i = tid; i < NL * DD; i += 512) {
            int nl = i >> 8, d = i & 255;
            sE[nl][d] = (float)Eb[((long)t * NN + n0 + nl) * DD + d];
        }
        __syncthreads();

        // pointer update: one thread per (nl, s)
        {
            int nl = tid >> 5, s = tid & 31;
            int n = n0 + nl;
            const bf16* mp = Mpb + ((long)n * SS + s) * DD;
            const bf16* mm = Mmb + ((long)n * SS + s) * DD;
            float dp = 0.f, dm = 0.f;
            for (int d = 0; d < DD; ++d) {
                float e = sE[nl][d];
                dp += e * (float)mp[d];
                dm += e * (float)mm[d];
            }
            float en = enorm[t * NN + n];
            float cp = dp / fmaxf(en * mpn[n * SS + s], EPS);
            float cm = dm / fmaxf(en * mmn[n * SS + s], EPS);
            sP[nl][s] += cp - cm;
        }
        __syncthreads();

        // memory read r = p @ M -> sA[:, :256] (bf16), obs -> sA[:, 256:384]
        for (int i = tid; i < NL * DD; i += 512) {
            int nl = i >> 8, d = i & 255;
            int n = n0 + nl;
            float acc = 0.f;
            for (int s = 0; s < SS; ++s)
                acc += sP[nl][s] * (float)Mb[((long)n * SS + s) * DD + d];
            sA[nl][d] = (bf16)acc;
        }
        for (int i = tid; i < NL * DOBS; i += 512) {
            int nl = i >> 7, k = i & 127;
            sA[nl][DD + k] = (bf16)obs[((long)t * NN + n0 + nl) * DOBS + k];
        }
        __syncthreads();

        // hid = relu([r|o] @ f_w + f_b): 16 waves, wave w -> cols [16w,16w+16)
        {
            v8f acc = {0.f, 0.f, 0.f, 0.f, 0.f, 0.f, 0.f, 0.f};
#pragma unroll
            for (int kk = 0; kk < KF / 32; ++kk) {
                v16bf a = load_tile_A(&sA[0][0] + kk * 32, KF, lane);
                v16bf b = load_tile_B_krow(fwb + (long)(kk * 32) * HH + wave * 16,
                                           HH, lane);
                acc = wmma_bf16(a, b, acc);
            }
            int col = wave * 16 + (lane & 15);
            int rb  = (lane >> 4) << 3;
#pragma unroll
            for (int j = 0; j < 8; ++j) {
                float h = fmaxf(acc[j] + fb[col], 0.f);
                sHid[rb + j][col] = h;
                sHb[rb + j][col]  = (bf16)h;
            }
        }
        __syncthreads();

        // logits = hid @ actor_w + actor_b: waves 0..3 (full waves -> EXEC ok)
        if (wave < 4) {
            v8f acc = {0.f, 0.f, 0.f, 0.f, 0.f, 0.f, 0.f, 0.f};
#pragma unroll
            for (int kk = 0; kk < HH / 32; ++kk) {
                v16bf a = load_tile_A(&sHb[0][0] + kk * 32, HH, lane);
                v16bf b = load_tile_B_krow(awb + (long)(kk * 32) * AA + wave * 16,
                                           AA, lane);
                acc = wmma_bf16(a, b, acc);
            }
            int col = wave * 16 + (lane & 15);
            int rb  = (lane >> 4) << 3;
#pragma unroll
            for (int j = 0; j < 8; ++j) sLog[rb + j][col] = acc[j] + ab[col];
        }
        __syncthreads();

        // critic, softmax, gumbel argmax: one thread per row
        if (tid < NL) {
            int nl = tid, n = n0 + nl;
            float v = cb[0];
            for (int dd = 0; dd < HH; ++dd) v += sHid[nl][dd] * cw[dd];
            float mx = sLog[nl][0];
            for (int c = 1; c < AA; ++c) mx = fmaxf(mx, sLog[nl][c]);
            float ssum = 0.f;
            for (int c = 0; c < AA; ++c) ssum += expf(sLog[nl][c] - mx);
            float inv = 1.f / ssum;
            long basei = (long)t * NN + n;
            int best = 0; float bestv = -3.4e38f;
            for (int c = 0; c < AA; ++c) {
                float lg = sLog[nl][c];
                probs_out[basei * AA + c] = expf(lg - mx) * inv;
                float uu = u[basei * AA + c];
                float g = -logf(-logf(uu + EPS) + EPS);
                float sc = lg + g;
                if (sc > bestv) { bestv = sc; best = c; }
            }
            a_out[basei] = best;
            v_out[basei] = v;
        }
        __syncthreads();
    }
}

// ---------------------------------------------------------------------------
// Host orchestration
// ---------------------------------------------------------------------------
extern "C" void kernel_launch(void* const* d_in, const int* in_sizes, int n_in,
                              void* d_out, int out_size, void* d_ws, size_t ws_size,
                              hipStream_t stream) {
    const float* obs    = (const float*)d_in[0];
    const int*   subt   = (const int*)d_in[1];
    const float* p_init = (const float*)d_in[2];
    const float* u      = (const float*)d_in[3];
    const float* embed  = (const float*)d_in[4];
    const float* W_ih   = (const float*)d_in[5];
    const float* W_hh   = (const float*)d_in[6];
    const float* b_ih   = (const float*)d_in[7];
    const float* b_hh   = (const float*)d_in[8];
    const float* psi_w  = (const float*)d_in[9];
    const float* psi_b  = (const float*)d_in[10];
    const float* f_w    = (const float*)d_in[11];
    const float* f_b    = (const float*)d_in[12];
    const float* cri_w  = (const float*)d_in[13];
    const float* cri_b  = (const float*)d_in[14];
    const float* act_w  = (const float*)d_in[15];
    const float* act_b  = (const float*)d_in[16];

    char* base = (char*)d_ws;
    size_t off = 0;
    auto alloc = [&](size_t bytes) -> void* {
        void* p = base + off;
        off = (off + bytes + 255) & ~(size_t)255;
        return p;
    };

    bf16* obs_bf = (bf16*)alloc((size_t)TT * NN * DOBS * 2);
    bf16* emb_bf = (bf16*)alloc((size_t)NN * SS * DE * 2);
    bf16* wih_bf = (bf16*)alloc((size_t)G3 * DE * 2);
    bf16* whh_bf = (bf16*)alloc((size_t)G3 * HID4 * 2);
    bf16* psw_bf = (bf16*)alloc((size_t)DOBS * DD * 2);
    bf16* fw_bf  = (bf16*)alloc((size_t)KF * HH * 2);
    bf16* aw_bf  = (bf16*)alloc((size_t)HH * AA * 2);
    float* h_f   = (float*)alloc((size_t)NN * HID4 * 4);
    bf16*  h_b   = (bf16*)alloc((size_t)NN * HID4 * 2);
    float* prz   = (float*)alloc((size_t)NN * 2048 * 4);
    float* gin   = (float*)alloc((size_t)NN * HID4 * 4);
    float* ghn   = (float*)alloc((size_t)NN * HID4 * 4);
    bf16* M_bf   = (bf16*)alloc((size_t)NN * SS * DD * 2);
    bf16* Mm_bf  = (bf16*)alloc((size_t)NN * SS * DD * 2);
    bf16* Mp_bf  = (bf16*)alloc((size_t)NN * SS * DD * 2);
    float* p0    = (float*)alloc((size_t)NN * SS * 4);
    float* mpn   = (float*)alloc((size_t)NN * SS * 4);
    float* mmn   = (float*)alloc((size_t)NN * SS * 4);
    bf16* E_bf   = (bf16*)alloc((size_t)TT * NN * DD * 2);
    float* enorm = (float*)alloc((size_t)TT * NN * 4);
    float* pbuf  = (float*)alloc((size_t)NN * SS * 4);

    // --- weight / input casts + embedding gather ---
    auto cast = [&](const float* s, bf16* d, long n) {
        cast_f32_bf16_kernel<<<(int)((n + 255) / 256), 256, 0, stream>>>(s, d, n);
    };
    cast(obs,   obs_bf, (long)TT * NN * DOBS);
    cast(W_ih,  wih_bf, (long)G3 * DE);
    cast(W_hh,  whh_bf, (long)G3 * HID4);
    cast(psi_w, psw_bf, (long)DOBS * DD);
    cast(f_w,   fw_bf,  (long)KF * HH);
    cast(act_w, aw_bf,  (long)HH * AA);
    gather_embed_kernel<<<(NN * SS * DE + 255) / 256, 256, 0, stream>>>(embed, subt,
                                                                        emb_bf);

    // --- GRU over S steps: async-LDS WMMA GEMM + gate fusion per step ---
    for (int s = 0; s < SS; ++s) {
        gru_gemm_kernel<<<(NN / BM) * (G3 / BN), 256, 0, stream>>>(
            emb_bf, h_b, wih_bf, whh_bf, prz, gin, ghn, s);
        gru_gate_kernel<<<(NN * HID4) / 256, 256, 0, stream>>>(
            prz, gin, ghn, b_ih, b_hh, h_f, h_b, M_bf, Mm_bf, Mp_bf, p0, s);
    }

    // --- psi projection (WMMA) + norms + p init ---
    psi_kernel<<<8192, 256, 0, stream>>>(obs_bf, psw_bf, psi_b, E_bf);
    rownorm_kernel<<<(TT * NN + 255) / 256, 256, 0, stream>>>(E_bf, enorm, TT * NN, DD);
    rownorm_kernel<<<(NN * SS + 255) / 256, 256, 0, stream>>>(Mp_bf, mpn, NN * SS, DD);
    rownorm_kernel<<<(NN * SS + 255) / 256, 256, 0, stream>>>(Mm_bf, mmn, NN * SS, DD);
    init_p_kernel<<<(NN + 255) / 256, 256, 0, stream>>>(p_init, p0, pbuf);

    // --- persistent T-scan with WMMA heads ---
    int*   a_out = (int*)d_out;
    float* probs = (float*)d_out + (size_t)TT * NN;
    float* v_out = (float*)d_out + (size_t)TT * NN + (size_t)TT * NN * AA;
    scan_kernel<<<NN / NL, 512, 0, stream>>>(E_bf, enorm, M_bf, Mp_bf, Mm_bf, mpn,
                                             mmn, pbuf, obs, u, fw_bf, f_b, aw_bf,
                                             act_b, cri_w, cri_b, a_out, probs,
                                             v_out);
}
